// GINModel_31628139167864
// MI455X (gfx1250) — matrix operand order repaired and checked by reading, not compiled, not run on hardware
//
#include <hip/hip_runtime.h>
#include <hip/hip_fp16.h>

#define NNODES 50000
#define NEDGES 800000
#define HDIM   96
#define ODIM   64

typedef __attribute__((ext_vector_type(2)))  float    v2f;
typedef __attribute__((ext_vector_type(8)))  float    v8f;
typedef __attribute__((ext_vector_type(16))) _Float16 v16h;

// ---------------------------------------------------------------------------
// h2 init: y = 2*x   (self-loop term + (1+eps)*h folded: h2 = 2h + sum_edges)
// ---------------------------------------------------------------------------
__global__ void k_double(const float* __restrict__ x, float* __restrict__ y, long n) {
    long i = (long)blockIdx.x * blockDim.x + threadIdx.x;
    long stride = (long)gridDim.x * blockDim.x;
    for (; i < n; i += stride) y[i] = 2.0f * x[i];
}

// ---------------------------------------------------------------------------
// Edge scatter-add: acc[dst[e]][c] += h[src[e]][c].  block = (96,4).
// L2-resident gather (h is 19.2MB << 192MB L2) + hw fp32 atomics.
// ---------------------------------------------------------------------------
__global__ void k_edges(const float* __restrict__ h, const int* __restrict__ src,
                        const int* __restrict__ dst, float* __restrict__ acc, int E) {
    int e = blockIdx.x * blockDim.y + threadIdx.y;
    if (e >= E) return;
    int c = threadIdx.x;
    long s = (long)src[e] * HDIM + c;
    long d = (long)dst[e] * HDIM + c;
    unsafeAtomicAdd(&acc[d], h[s]);   // global_atomic_add_f32, no return
}

// ---------------------------------------------------------------------------
// WMMA GEMM: C[M,NCOLS] = act(A[M,96] @ W[96,NCOLS] + bias)
// One wave = one 16x16 output tile. 8 waves / block; waves of a block cover
// consecutive n-tiles of the same m-tile rows for A reuse through L1.
// ---------------------------------------------------------------------------
template<int NCOLS, bool RELU>
__global__ __launch_bounds__(256) void k_gemm(const float* __restrict__ A,
                                              const float* __restrict__ W,
                                              const float* __restrict__ bias,
                                              float* __restrict__ C, int M) {
    constexpr int NT = NCOLS / 16;        // n tiles
    const int wave = threadIdx.x >> 5;
    const int lane = threadIdx.x & 31;
    const int gw   = blockIdx.x * 8 + wave;
    const int tm   = gw / NT;
    const int tn   = gw % NT;
    const int mtiles = M >> 4;            // M = 50000, divisible by 16
    if (tm >= mtiles) return;             // wave-uniform: EXEC stays all-ones

    const int m  = lane & 15;             // row within A-tile / col within B-tile
    const int kh = lane >> 4;             // K half selector
    const float* __restrict__ arow = A + (long)(tm * 16 + m) * HDIM;
    const int col = tn * 16 + m;

    v8f acc = {};
#if __has_builtin(__builtin_amdgcn_wmma_f32_16x16x4_f32)
    // Full-precision fp32 WMMA: A 16x4, B 4x16, 24 steps over K=96.
    #pragma unroll
    for (int k0 = 0; k0 < HDIM; k0 += 4) {
        v2f a, b;
        a.x = arow[k0 + 2 * kh];
        a.y = arow[k0 + 2 * kh + 1];
        b.x = W[(long)(k0 + 2 * kh)     * NCOLS + col];
        b.y = W[(long)(k0 + 2 * kh + 1) * NCOLS + col];
        acc = __builtin_amdgcn_wmma_f32_16x16x4_f32(
                  false, a, false, b, (short)0, acc, false, false);
    }
#else
    // Fallback: f16 inputs, fp32 accumulate (codegen-verified builtin).
    #pragma unroll
    for (int k0 = 0; k0 < HDIM; k0 += 32) {
        v16h a, b;
        #pragma unroll
        for (int j = 0; j < 16; ++j)
            a[j] = (_Float16)arow[k0 + 16 * kh + j];
        #pragma unroll
        for (int j = 0; j < 16; ++j)
            b[j] = (_Float16)W[(long)(k0 + 16 * kh + j) * NCOLS + col];
        acc = __builtin_amdgcn_wmma_f32_16x16x32_f16(
                  false, a, false, b, (short)0, acc, false, false);
    }
#endif

    const float bcol = bias[col];
    #pragma unroll
    for (int j = 0; j < 8; ++j) {
        int row = tm * 16 + j + 8 * kh;   // C layout: VGPR j -> M = j + 8*kh
        float v = acc[j] + bcol;
        if (RELU) v = fmaxf(v, 0.0f);
        C[(long)row * NCOLS + col] = v;
    }
}

// ---------------------------------------------------------------------------
// BatchNorm stats (training mode, biased variance). One block per column,
// deterministic LDS tree reduction; folds gamma/beta into scale/shift.
// ---------------------------------------------------------------------------
__global__ __launch_bounds__(256) void k_bnstats(const float* __restrict__ z,
                                                 const float* __restrict__ gamma,
                                                 const float* __restrict__ beta,
                                                 float* __restrict__ scale,
                                                 float* __restrict__ shift, int M) {
    __shared__ float ssum[256];
    __shared__ float ssq[256];
    const int c = blockIdx.x;
    float s = 0.0f, q = 0.0f;
    for (int r = threadIdx.x; r < M; r += 256) {
        float v = z[(long)r * HDIM + c];
        s += v;
        q += v * v;
    }
    ssum[threadIdx.x] = s;
    ssq[threadIdx.x]  = q;
    __syncthreads();
    for (int off = 128; off > 0; off >>= 1) {
        if (threadIdx.x < off) {
            ssum[threadIdx.x] += ssum[threadIdx.x + off];
            ssq[threadIdx.x]  += ssq[threadIdx.x + off];
        }
        __syncthreads();
    }
    if (threadIdx.x == 0) {
        float mu  = ssum[0] / (float)M;
        float var = ssq[0] / (float)M - mu * mu;
        float sc  = gamma[c] / sqrtf(var + 1e-5f);
        scale[c] = sc;
        shift[c] = beta[c] - mu * sc;
    }
}

// ---------------------------------------------------------------------------
// BN apply + ReLU: h = max(z*scale[c] + shift[c], 0)
// ---------------------------------------------------------------------------
__global__ void k_bnapply(const float* __restrict__ z, const float* __restrict__ scale,
                          const float* __restrict__ shift, float* __restrict__ h, long n) {
    long i = (long)blockIdx.x * blockDim.x + threadIdx.x;
    long stride = (long)gridDim.x * blockDim.x;
    for (; i < n; i += stride) {
        int c = (int)(i % HDIM);
        h[i] = fmaxf(fmaf(z[i], scale[c], shift[c]), 0.0f);
    }
}

// ---------------------------------------------------------------------------
extern "C" void kernel_launch(void* const* d_in, const int* in_sizes, int n_in,
                              void* d_out, int out_size, void* d_ws, size_t ws_size,
                              hipStream_t stream) {
    const float* features = (const float*)d_in[0];
    const float* W0a = (const float*)d_in[1];
    const float* b0a = (const float*)d_in[2];
    const float* W0b = (const float*)d_in[3];
    const float* b0b = (const float*)d_in[4];
    const float* W1a = (const float*)d_in[5];
    const float* b1a = (const float*)d_in[6];
    const float* W1b = (const float*)d_in[7];
    const float* b1b = (const float*)d_in[8];
    const float* gamma0 = (const float*)d_in[9];
    const float* beta0  = (const float*)d_in[10];
    const float* gamma1 = (const float*)d_in[11];
    const float* beta1  = (const float*)d_in[12];
    const float* Wp = (const float*)d_in[13];
    const float* bp = (const float*)d_in[14];
    const int* src = (const int*)d_in[15];
    const int* dst = (const int*)d_in[16];
    float* out = (float*)d_out;

    float* buf0  = (float*)d_ws;                        // [N, 96]
    float* buf1  = buf0 + (size_t)NNODES * HDIM;        // [N, 96]
    float* scale = buf1 + (size_t)NNODES * HDIM;        // [96]
    float* shift = scale + HDIM;                        // [96]

    const long nelem = (long)NNODES * HDIM;             // 4.8M
    const int eg = (NEDGES + 3) / 4;                    // edge blocks
    const dim3 eb(HDIM, 4);
    const int g_gemm96 = ((NNODES / 16) * (HDIM / 16) + 7) / 8;  // 2344
    const int g_gemm64 = ((NNODES / 16) * (ODIM / 16) + 7) / 8;  // 1563
    const int g_elem = 18750;

    // ---- layer 0 ----
    k_double<<<g_elem, 256, 0, stream>>>(features, buf0, nelem);
    k_edges<<<eg, eb, 0, stream>>>(features, src, dst, buf0, NEDGES);
    k_gemm<HDIM, true ><<<g_gemm96, 256, 0, stream>>>(buf0, W0a, b0a, buf1, NNODES);
    k_gemm<HDIM, false><<<g_gemm96, 256, 0, stream>>>(buf1, W0b, b0b, buf0, NNODES);
    k_bnstats<<<HDIM, 256, 0, stream>>>(buf0, gamma0, beta0, scale, shift, NNODES);
    k_bnapply<<<g_elem, 256, 0, stream>>>(buf0, scale, shift, buf1, nelem);

    // ---- layer 1 ----
    k_double<<<g_elem, 256, 0, stream>>>(buf1, buf0, nelem);
    k_edges<<<eg, eb, 0, stream>>>(buf1, src, dst, buf0, NEDGES);
    k_gemm<HDIM, true ><<<g_gemm96, 256, 0, stream>>>(buf0, W1a, b1a, buf1, NNODES);
    k_gemm<HDIM, false><<<g_gemm96, 256, 0, stream>>>(buf1, W1b, b1b, buf0, NNODES);
    k_bnstats<<<HDIM, 256, 0, stream>>>(buf0, gamma1, beta1, scale, shift, NNODES);
    k_bnapply<<<g_elem, 256, 0, stream>>>(buf0, scale, shift, buf1, nelem);

    // ---- prediction head ----
    k_gemm<ODIM, false><<<g_gemm64, 256, 0, stream>>>(buf1, Wp, bp, out, NNODES);
}
